// GeometricEmbedding_6253472382976
// MI455X (gfx1250) — matrix operand order repaired and checked by reading, not compiled, not run on hardware
//
#include <hip/hip_runtime.h>
#include <math.h>

typedef __attribute__((ext_vector_type(16))) _Float16 v16h;
typedef __attribute__((ext_vector_type(8)))  _Float16 v8h;
typedef __attribute__((ext_vector_type(8)))  float    v8f;

#define DFEAT 64
#define NGRAPH 64
#define WT_STRIDE 80   // padded stride (halves) for transposed W in LDS; multiple of 16 -> 32B aligned rows

// ---------------- utility ----------------
__global__ void fill_kernel(float* __restrict__ p, float v, int n) {
    int t = blockIdx.x * blockDim.x + threadIdx.x;
    if (t < n) p[t] = v;
}

// deg[col[e]] += 1  (deg pre-filled with 1.0 for the self loop)
__global__ void deg_accum_kernel(const int* __restrict__ col, float* __restrict__ deg, int E) {
    int e = blockIdx.x * blockDim.x + threadIdx.x;
    if (e < E) atomicAdd(&deg[col[e]], 1.0f);
}

__global__ void dinv_kernel(const float* __restrict__ deg, float* __restrict__ dinv, int n) {
    int t = blockIdx.x * blockDim.x + threadIdx.x;
    if (t < n) {
        float d = deg[t];
        dinv[t] = d > 0.0f ? rsqrtf(d) : 0.0f;
    }
}

__global__ void norm_kernel(const int* __restrict__ row, const int* __restrict__ col,
                            const float* __restrict__ dinv, float* __restrict__ nrm, int E) {
    int e = blockIdx.x * blockDim.x + threadIdx.x;
    if (e < E) nrm[e] = dinv[row[e]] * dinv[col[e]];
}

// X[n][:] = emb_table[node_types[n]][:]   (16 threads per node, float4 each)
__global__ void embed_kernel(const int* __restrict__ types, const float* __restrict__ emb,
                             float* __restrict__ X, int N) {
    int t = blockIdx.x * blockDim.x + threadIdx.x;
    if (t >= N * 16) return;
    int n = t >> 4, sub = t & 15;
    float4 v = ((const float4*)(emb + (size_t)types[n] * DFEAT))[sub];
    ((float4*)(X + (size_t)n * DFEAT))[sub] = v;
}

// ---------------- WMMA GEMM:  H[N][64] = X[N][64] @ W[64][64]  (f16 compute, f32 acc)
// Block = 128 threads (4 waves). Each wave computes a 16x64 output strip.
// W is staged TRANSPOSED in LDS so each B fragment is one contiguous 32-byte load.
__global__ __launch_bounds__(128) void gemm64_wmma(const float* __restrict__ X,
                                                   const float* __restrict__ W,
                                                   float* __restrict__ H, int N) {
    __shared__ __align__(16) _Float16 Wt[DFEAT * WT_STRIDE]; // 10 KB, Wt[n][k] = W[k][n]
    __shared__ __align__(16) _Float16 Xh[64 * DFEAT];        // 8 KB

    const int tid  = threadIdx.x;
    const int lane = tid & 31;
    const int wave = tid >> 5;          // 0..3
    const int rowBase = blockIdx.x * 64;

    // stage W transposed (coalesced global reads, scattered LDS writes)
    for (int i = tid; i < DFEAT * DFEAT; i += 128) {
        int k = i >> 6, n = i & 63;
        Wt[n * WT_STRIDE + k] = (_Float16)W[i];
    }
    // stage a 64-row X tile (coalesced)
    for (int i = tid; i < 64 * DFEAT; i += 128) {
        int r = i >> 6, c = i & 63;
        int gr = rowBase + r;
        Xh[i] = (_Float16)(gr < N ? X[(size_t)gr * DFEAT + c] : 0.0f);
    }
    __syncthreads();

    // CDNA5 wave32 WMMA operand layouts as used here (A/B in contiguous k order, not the ISA table order):
    // A (16x32 f16): lanes 0-15 -> M=lane, K base 0 ; lanes 16-31 -> K base 16
    //                half h -> K = kbaseA + h   (two contiguous 8-half runs)
    // B (32x16 f16): lanes 0-15 -> K = 0..15 ; lanes 16-31 -> K = 16..31 ; N = lane&15
    //                -> with Wt[n][k], one contiguous 16-half (32 B) run per fragment
    // C/D (16x16 f32): VGPR r -> M = r (lanes 0-15) / r+8 (lanes 16-31), N = lane&15
    const int m      = (wave << 4) | (lane & 15);
    const int kbaseA = (lane < 16) ? 0 : 16;
    const int nB     = lane & 15;
    const int kbaseB = (lane < 16) ? 0 : 16;

    v8f acc[4] = {v8f{}, v8f{}, v8f{}, v8f{}};

    #pragma unroll
    for (int ks = 0; ks < 2; ++ks) {
        const int koff = ks * 32;
        // A fragment: two contiguous 8-half runs -> 16B LDS loads
        v8h alo = *(const v8h*)(Xh + m * DFEAT + koff + kbaseA);
        v8h ahi = *(const v8h*)(Xh + m * DFEAT + koff + kbaseA + 8);
        v16h a;
        #pragma unroll
        for (int h = 0; h < 8; ++h) { a[h] = alo[h]; a[h + 8] = ahi[h]; }

        #pragma unroll
        for (int nt = 0; nt < 4; ++nt) {
            // one 32-byte contiguous LDS load per B fragment
            v16h b = *(const v16h*)(Wt + ((nt << 4) + nB) * WT_STRIDE + koff + kbaseB);
            acc[nt] = __builtin_amdgcn_wmma_f32_16x16x32_f16(
                false, a, false, b, (short)0, acc[nt], false, false);
        }
    }

    const int mlo = (lane < 16) ? 0 : 8;
    #pragma unroll
    for (int r = 0; r < 8; ++r) {
        int grow = rowBase + (wave << 4) + mlo + r;
        if (grow < N) {
            float* dst = H + (size_t)grow * DFEAT;
            dst[nB]      = acc[0][r];
            dst[16 + nB] = acc[1][r];
            dst[32 + nB] = acc[2][r];
            dst[48 + nB] = acc[3][r];
        }
    }
}

// ---------------- edge scatter: OUT[col[e]] += norm[e] * H[row[e]]
// 16 threads per edge, float4 gather, 4 f32 atomic adds each (hits L2)
__global__ void scatter_kernel(const int* __restrict__ row, const int* __restrict__ col,
                               const float* __restrict__ nrm, const float* __restrict__ H,
                               float* __restrict__ OUT, int E) {
    long long t = (long long)blockIdx.x * blockDim.x + threadIdx.x;
    long long e = t >> 4;
    if (e >= E) return;
    int sub = (int)(t & 15);
    int r = row[e], c = col[e];
    float w = nrm[e];
    float4 hv = ((const float4*)(H + (size_t)r * DFEAT))[sub];
    float* dst = OUT + (size_t)c * DFEAT + sub * 4;
    atomicAdd(dst + 0, w * hv.x);
    atomicAdd(dst + 1, w * hv.y);
    atomicAdd(dst + 2, w * hv.z);
    atomicAdd(dst + 3, w * hv.w);
}

// x_next = [relu]( OUT + dinv^2 * H (self loop) + b )
__global__ void finish_kernel(const float* __restrict__ OUT, const float* __restrict__ H,
                              const float* __restrict__ dinv, const float* __restrict__ b,
                              float* __restrict__ Xn, int N, int doRelu) {
    int t = blockIdx.x * blockDim.x + threadIdx.x;
    if (t >= N * DFEAT) return;
    int n = t >> 6, d = t & 63;
    float di = dinv[n];
    float v = OUT[t] + di * di * H[t] + b[d];
    if (doRelu) v = fmaxf(v, 0.0f);
    Xn[t] = v;
}

// ---------------- pooling ----------------
__global__ void pool_accum_kernel(const float* __restrict__ X, const int* __restrict__ batch,
                                  float* __restrict__ sums, float* __restrict__ cnts, int N) {
    int t = blockIdx.x * blockDim.x + threadIdx.x;
    if (t >= N * DFEAT) return;
    int n = t >> 6, d = t & 63;
    int g = batch[n];
    atomicAdd(&sums[g * DFEAT + d], X[t]);
    if (d == 0) atomicAdd(&cnts[g], 1.0f);
}

__global__ void pool_finalize_kernel(const float* __restrict__ sums,
                                     const float* __restrict__ cnts,
                                     float* __restrict__ out) {
    __shared__ float ssq[DFEAT];
    int g = blockIdx.x;     // 64 graphs
    int d = threadIdx.x;    // 64 features
    float c = fmaxf(cnts[g], 1.0f);
    float m = sums[g * DFEAT + d] / c;
    ssq[d] = m * m;
    __syncthreads();
    for (int s = 32; s > 0; s >>= 1) {
        if (d < s) ssq[d] += ssq[d + s];
        __syncthreads();
    }
    out[g * DFEAT + d] = m / sqrtf(ssq[0]);
}

// ---------------- driver ----------------
extern "C" void kernel_launch(void* const* d_in, const int* in_sizes, int n_in,
                              void* d_out, int out_size, void* d_ws, size_t ws_size,
                              hipStream_t stream) {
    const int*   node_types = (const int*)d_in[0];
    const int*   edge_index = (const int*)d_in[1];
    const int*   batch      = (const int*)d_in[2];
    const float* emb        = (const float*)d_in[3];
    const float* Wmat[3]    = {(const float*)d_in[4], (const float*)d_in[6], (const float*)d_in[8]};
    const float* bvec[3]    = {(const float*)d_in[5], (const float*)d_in[7], (const float*)d_in[9]};

    const int N = in_sizes[0];
    const int E = in_sizes[1] / 2;
    const int* row = edge_index;        // edge_index[0,:]
    const int* col = edge_index + E;    // edge_index[1,:]

    float* ws = (float*)d_ws;
    float* deg  = ws;  ws += N;
    float* dinv = ws;  ws += N;
    float* nrm  = ws;  ws += E;
    float* bufA = ws;  ws += (size_t)N * DFEAT;   // current features x
    float* bufB = ws;  ws += (size_t)N * DFEAT;   // h = x @ W
    float* bufC = ws;  ws += (size_t)N * DFEAT;   // scatter accumulator
    float* sums = ws;  ws += NGRAPH * DFEAT;
    float* cnts = ws;  ws += NGRAPH;

    const int T = 256;
    auto blk = [](long long n, int t) { return (unsigned)((n + t - 1) / t); };

    // degree (self loop counted as 1.0 init) -> dinv -> per-edge norm
    fill_kernel<<<blk(N, T), T, 0, stream>>>(deg, 1.0f, N);
    deg_accum_kernel<<<blk(E, T), T, 0, stream>>>(col, deg, E);
    dinv_kernel<<<blk(N, T), T, 0, stream>>>(deg, dinv, N);
    norm_kernel<<<blk(E, T), T, 0, stream>>>(row, col, dinv, nrm, E);

    // x = emb_table[node_types]
    embed_kernel<<<blk((long long)N * 16, T), T, 0, stream>>>(node_types, emb, bufA, N);

    // three GCN layers
    for (int l = 0; l < 3; ++l) {
        gemm64_wmma<<<blk(N, 64), 128, 0, stream>>>(bufA, Wmat[l], bufB, N);
        fill_kernel<<<blk((long long)N * DFEAT, T), T, 0, stream>>>(bufC, 0.0f, N * DFEAT);
        scatter_kernel<<<blk((long long)E * 16, T), T, 0, stream>>>(row, col, nrm, bufB, bufC, E);
        finish_kernel<<<blk((long long)N * DFEAT, T), T, 0, stream>>>(
            bufC, bufB, dinv, bvec[l], bufA, N, (l < 2) ? 1 : 0);
    }

    // mean pool per graph + L2 normalize rows
    fill_kernel<<<blk(NGRAPH * DFEAT + NGRAPH, T), T, 0, stream>>>(sums, 0.0f, NGRAPH * DFEAT + NGRAPH);
    pool_accum_kernel<<<blk((long long)N * DFEAT, T), T, 0, stream>>>(bufA, batch, sums, cnts, N);
    pool_finalize_kernel<<<NGRAPH, DFEAT, 0, stream>>>(sums, cnts, (float*)d_out);
}
